// EdgeAttentionFlat_23991687315682
// MI455X (gfx1250) — matrix operand order repaired and checked by reading, not compiled, not run on hardware
//
#include <hip/hip_runtime.h>
#include <hip/hip_bf16.h>
#include <math.h>

// EdgeAttention for MI455X (gfx1250, wave32, WMMA).
// All matmuls mapped to v_wmma_f32_16x16x32_f16 (f16 in, f32 accum).
// Working set ~135MB -> resident in 192MB L2; kernels chained on one stream.
// K/V tile staging uses the CDNA5 async global->LDS copy path (ASYNCcnt).
//
// Shapes: B=1, N=128, D=256, H=8, DK=32.

#define Nn   128
#define Dd   256
#define Hh   8
#define DK   32
#define ROWS (Nn * Nn)           // 16384 flattened (x,y) rows

typedef __attribute__((ext_vector_type(16))) _Float16 v16h;
typedef __attribute__((ext_vector_type(8)))  float    v8f;

#define WMMA(a, b, c) __builtin_amdgcn_wmma_f32_16x16x32_f16( \
    false, (a), false, (b), (short)0, (c), false, false)

// ---------- fragment loaders (CDNA5 wave32 layouts, CDNA5 ISA §7.12.2) ----

// A-matrix 16x32 f16, row-major source with stride ld.
// lane<16: halves 0..7 -> K=0..7, halves 8..15 -> K=16..23 ; lane>=16: +8.
__device__ inline v16h load_a_f16(const _Float16* __restrict__ src, int ld) {
  const int lane = threadIdx.x & 31;
  const int m  = lane & 15;
  const int kb = (lane >> 4) << 3;          // 0 or 8
  const _Float16* row = src + m * ld;
  v16h a;
#pragma unroll
  for (int j = 0; j < 8; ++j) a[j]     = row[kb + j];
#pragma unroll
  for (int j = 0; j < 8; ++j) a[8 + j] = row[16 + kb + j];
  return a;
}

// Same, converting from f32 source.
__device__ inline v16h load_a_f32(const float* __restrict__ src, int ld) {
  const int lane = threadIdx.x & 31;
  const int m  = lane & 15;
  const int kb = (lane >> 4) << 3;
  const float* row = src + m * ld;
  v16h a;
#pragma unroll
  for (int j = 0; j < 8; ++j) a[j]     = (_Float16)row[kb + j];
#pragma unroll
  for (int j = 0; j < 8; ++j) a[8 + j] = (_Float16)row[16 + kb + j];
  return a;
}

// B-matrix 32x16 where B = M^T and M is row-major 16(outcols) x 32(K):
// column n of B == row n of M (contiguous). lane half selects K half.
__device__ inline v16h load_bt_f16(const _Float16* __restrict__ Msrc, int ld) {
  const int lane = threadIdx.x & 31;
  const int n  = lane & 15;
  const int kh = (lane >> 4) << 3;          // 0 or 8
  const _Float16* row = Msrc + n * ld;
  v16h b;
#pragma unroll
  for (int j = 0; j < 16; ++j) b[j] = row[kh + j + (j & 8)];
  return b;
}

__device__ inline v16h load_bt_f32(const float* __restrict__ Msrc, int ld) {
  const int lane = threadIdx.x & 31;
  const int n  = lane & 15;
  const int kh = (lane >> 4) << 3;
  const float* row = Msrc + n * ld;
  v16h b;
#pragma unroll
  for (int j = 0; j < 16; ++j) b[j] = (_Float16)row[kh + j + (j & 8)];
  return b;
}

// B-matrix 32x16 from plain row-major K x N storage (stride ld).
__device__ inline v16h load_b_f16(const _Float16* __restrict__ src, int ld) {
  const int lane = threadIdx.x & 31;
  const int n  = lane & 15;
  const int kh = (lane >> 4) << 3;
  v16h b;
#pragma unroll
  for (int j = 0; j < 16; ++j) b[j] = src[(kh + j + (j & 8)) * ld + n];
  return b;
}

// C/D 16x16 f32: VGPR i, lanes<16 -> M=i, lanes>=16 -> M=8+i; N = lane&15.
__device__ inline void store_c_f16(_Float16* __restrict__ dst, int ld, v8f c) {
  const int lane = threadIdx.x & 31;
  const int n  = lane & 15;
  const int mb = (lane >> 4) << 3;
#pragma unroll
  for (int i = 0; i < 8; ++i) dst[(mb + i) * ld + n] = (_Float16)c[i];
}

// ---------- async global->LDS staging of a 128x32 f16 tile (8 KB) -----------
// CDNA5 path: GLOBAL_LOAD_ASYNC_TO_LDS_B128 deposits straight into LDS with no
// VGPR landing; tracked by ASYNCcnt. The flat address of a __shared__ object
// carries the raw LDS byte offset in its low 32 bits (LDS aperture, ISA 10.2),
// which is exactly the VDST operand this instruction wants. INST_OFFSET is
// added to BOTH the LDS and global addresses, so one address setup covers both
// 16-byte halves of each thread's 32-byte slice.
__device__ inline void async_fill_tile_128x32(const _Float16* __restrict__ g,
                                              int g_ld, _Float16* lds_base) {
  const int t = threadIdx.x;               // 256 threads, 32B each
  const int row = t >> 1, part = (t & 1) * 16;
  const _Float16* src = g + (size_t)row * g_ld + part;
  const unsigned lds_off = (unsigned)(uintptr_t)(lds_base + row * DK + part);
  asm volatile(
      "global_load_async_to_lds_b128 %0, %1, off\n\t"
      "global_load_async_to_lds_b128 %0, %1, off offset:16"
      :: "v"(lds_off), "v"(src) : "memory");
  asm volatile("s_wait_asynccnt 0" ::: "memory");
}

// ---------------- kernel 1: projection  P = X @ W^T  (f32 -> f16 x2 layouts)
// grid.x = ROWS/16 row tiles; 256 threads = 8 waves, wave w -> cols [32w,32w+32)
__global__ void proj_qkv_kernel(const float* __restrict__ X,
                                const float* __restrict__ W,
                                _Float16* __restrict__ P,    // [x][y][c]
                                _Float16* __restrict__ PT) { // [y][x][c]
  const int r0   = blockIdx.x * 16;
  const int wave = threadIdx.x >> 5;
  const int lane = threadIdx.x & 31;
  const int n0   = wave * 32;
  v8f acc0 = {}, acc1 = {};
#pragma unroll
  for (int kk = 0; kk < 8; ++kk) {
    const int k0 = kk * 32;
    v16h a  = load_a_f32(X + (size_t)r0 * Dd + k0, Dd);
    v16h b0 = load_bt_f32(W + (size_t)(n0     ) * Dd + k0, Dd);
    v16h b1 = load_bt_f32(W + (size_t)(n0 + 16) * Dd + k0, Dd);
    acc0 = WMMA(a, b0, acc0);
    acc1 = WMMA(a, b1, acc1);
  }
  const int n  = lane & 15;
  const int mb = (lane >> 4) << 3;
  const int x  = r0 >> 7;                   // tile never crosses an x boundary
#pragma unroll
  for (int t = 0; t < 2; ++t) {
    const v8f acc = t ? acc1 : acc0;
    const int c = n0 + t * 16 + n;
#pragma unroll
    for (int i = 0; i < 8; ++i) {
      const int r = r0 + mb + i;
      const _Float16 hv = (_Float16)acc[i];
      P [(size_t)r * Dd + c] = hv;
      PT[((size_t)(r & 127) * Nn + x) * Dd + c] = hv;   // transposed layout
    }
  }
}

// ---------------- kernel 2: masked scores  S = scale * Q K^T  (per (p,h))
// mode 0: p = x (row attention), mode 1: p = y on transposed tensors.
// K tile (128x32 f16, 8KB) staged in LDS via async copy, reused by all 8 waves.
__global__ void scores_kernel(const _Float16* __restrict__ Q,
                              const _Float16* __restrict__ K,
                              const unsigned char* __restrict__ mask, // [x][y][z]
                              _Float16* __restrict__ S, int mode) {
  __shared__ _Float16 kt[Nn * DK];
  const int p = blockIdx.x, h = blockIdx.y;
  const _Float16* qb = Q + ((size_t)p * Nn) * Dd + h * DK;
  const _Float16* kg = K + ((size_t)p * Nn) * Dd + h * DK;
  async_fill_tile_128x32(kg, Dd, kt);
  __syncthreads();
  const int wave = threadIdx.x >> 5, lane = threadIdx.x & 31;
  const int m0 = wave * 16;
  const int n  = lane & 15, mb = (lane >> 4) << 3;
  const float scale = 0.17677669529663687f;           // 1/sqrt(DK)
  const _Float16 negbig = (_Float16)(-30000.0f);
  _Float16* sb = S + (size_t)(p * Hh + h) * (Nn * Nn);
  v16h a = load_a_f16(qb + (size_t)m0 * Dd, Dd);      // K-dim = 32: single step
#pragma unroll
  for (int nt = 0; nt < 8; ++nt) {
    v16h b = load_bt_f16(kt + nt * 16 * DK, DK);
    v8f c = {};
    c = WMMA(a, b, c);
    const int z = nt * 16 + n;
#pragma unroll
    for (int i = 0; i < 8; ++i) {
      const int row = m0 + mb + i;                    // y (mode0) or x (mode1)
      const size_t mi = (mode == 0)
          ? ((size_t)(p   * Nn + row) * Nn + z)       // mask[x=p ][y=row][z]
          : ((size_t)(row * Nn + p  ) * Nn + z);      // mask[x=row][y=p ][z]
      sb[(size_t)row * Nn + z] = mask[mi] ? negbig : (_Float16)(c[i] * scale);
    }
  }
}

// ---------------- kernel 3: joint softmax over concat(r,l) (256 elems), inplace
// one wave32 per (x,y,h); each lane owns 4 r-elems + 4 l-elems.
__global__ void softmax_kernel(_Float16* __restrict__ Sr, _Float16* __restrict__ Sl) {
  const int wave = threadIdx.x >> 5, lane = threadIdx.x & 31;
  const int g = blockIdx.x * 8 + wave;                 // ((x*128)+y)*8 + h
  const int h = g & 7, y = (g >> 3) & 127, x = g >> 10;
  _Float16* vr = Sr + (size_t)(x * Hh + h) * (Nn * Nn) + (size_t)y * Nn;
  _Float16* vl = Sl + (size_t)(y * Hh + h) * (Nn * Nn) + (size_t)x * Nn;
  float r[4], l[4];
#pragma unroll
  for (int j = 0; j < 4; ++j) {
    r[j] = (float)vr[lane * 4 + j];
    l[j] = (float)vl[lane * 4 + j];
  }
  float m = -3.0e38f;
#pragma unroll
  for (int j = 0; j < 4; ++j) m = fmaxf(m, fmaxf(r[j], l[j]));
#pragma unroll
  for (int off = 16; off > 0; off >>= 1) m = fmaxf(m, __shfl_xor(m, off, 32));
  float s = 0.0f;
#pragma unroll
  for (int j = 0; j < 4; ++j) {
    r[j] = expf(r[j] - m);  l[j] = expf(l[j] - m);  s += r[j] + l[j];
  }
#pragma unroll
  for (int off = 16; off > 0; off >>= 1) s += __shfl_xor(s, off, 32);
  const float inv = 1.0f / s;
#pragma unroll
  for (int j = 0; j < 4; ++j) {
    vr[lane * 4 + j] = (_Float16)(r[j] * inv);
    vl[lane * 4 + j] = (_Float16)(l[j] * inv);
  }
}

// ---------------- kernel 4: x = att @ V  (128x32 <- 128x128 x 128x32), per (p,h)
__global__ void av_kernel(const _Float16* __restrict__ A,   // att [p][h][row][z]
                          const _Float16* __restrict__ V,   // [p*Nn + z][h*DK + d]
                          _Float16* __restrict__ O) {       // [p*Nn + row][h*DK + d]
  __shared__ _Float16 vt[Nn * DK];
  const int p = blockIdx.x, h = blockIdx.y;
  const _Float16* ab = A + (size_t)(p * Hh + h) * (Nn * Nn);
  const _Float16* vg = V + ((size_t)p * Nn) * Dd + h * DK;
  _Float16* ob = O + ((size_t)p * Nn) * Dd + h * DK;
  async_fill_tile_128x32(vg, Dd, vt);
  __syncthreads();
  const int wave = threadIdx.x >> 5;
  const int m0 = wave * 16;
  v8f acc0 = {}, acc1 = {};
#pragma unroll
  for (int kt = 0; kt < 4; ++kt) {
    v16h a  = load_a_f16(ab + (size_t)m0 * Nn + kt * 32, Nn);
    v16h b0 = load_b_f16(vt + kt * 32 * DK +  0, DK);
    v16h b1 = load_b_f16(vt + kt * 32 * DK + 16, DK);
    acc0 = WMMA(a, b0, acc0);
    acc1 = WMMA(a, b1, acc1);
  }
  store_c_f16(ob + (size_t)m0 * Dd +  0, Dd, acc0);
  store_c_f16(ob + (size_t)m0 * Dd + 16, Dd, acc1);
}

// ---------------- kernel 5: OUT = (x_r + x_l^T-layout) @ Wo^T  (f32 output)
__global__ void out_kernel(const _Float16* __restrict__ XR,  // [x][y][c]
                           const _Float16* __restrict__ XL,  // [y][x][c]
                           const float* __restrict__ Wo,
                           float* __restrict__ OUT) {
  const int r0   = blockIdx.x * 16;
  const int wave = threadIdx.x >> 5;
  const int lane = threadIdx.x & 31;
  const int x = r0 >> 7, y0 = r0 & 127;
  const int n0 = wave * 32;
  const int m  = lane & 15;
  const int kb = (lane >> 4) << 3;
  v8f acc0 = {}, acc1 = {};
#pragma unroll
  for (int kk = 0; kk < 8; ++kk) {
    const int k0 = kk * 32;
    const _Float16* rr = XR + (size_t)(r0 + m) * Dd + k0;
    const _Float16* rl = XL + ((size_t)(y0 + m) * Nn + x) * Dd + k0;
    v16h a;
#pragma unroll
    for (int j = 0; j < 8; ++j) a[j]     = rr[kb + j]      + rl[kb + j];
#pragma unroll
    for (int j = 0; j < 8; ++j) a[8 + j] = rr[16 + kb + j] + rl[16 + kb + j];
    v16h b0 = load_bt_f32(Wo + (size_t)(n0     ) * Dd + k0, Dd);
    v16h b1 = load_bt_f32(Wo + (size_t)(n0 + 16) * Dd + k0, Dd);
    acc0 = WMMA(a, b0, acc0);
    acc1 = WMMA(a, b1, acc1);
  }
  const int n  = lane & 15;
  const int mb = (lane >> 4) << 3;
#pragma unroll
  for (int t = 0; t < 2; ++t) {
    const v8f acc = t ? acc1 : acc0;
#pragma unroll
    for (int i = 0; i < 8; ++i)
      OUT[(size_t)(r0 + mb + i) * Dd + n0 + t * 16 + n] = acc[i];
  }
}

extern "C" void kernel_launch(void* const* d_in, const int* in_sizes, int n_in,
                              void* d_out, int out_size, void* d_ws, size_t ws_size,
                              hipStream_t stream) {
  const float* query = (const float*)d_in[0];
  const float* key   = (const float*)d_in[1];
  const float* value = (const float*)d_in[2];
  const unsigned char* mask = (const unsigned char*)d_in[3]; // jnp.bool_ = 1 byte
  const float* Wq = (const float*)d_in[4];
  const float* Wk = (const float*)d_in[5];
  const float* Wv = (const float*)d_in[6];
  const float* Wo = (const float*)d_in[7];
  float* out = (float*)d_out;

  // workspace: 128 MB total, all f16
  _Float16* w = (_Float16*)d_ws;
  const size_t SZP = (size_t)ROWS * Dd;               // 16384*256 halves
  const size_t SSZ = (size_t)Nn * Hh * Nn * Nn;       // score tensor halves
  _Float16* q16 = w;
  _Float16* k16 = w + 1 * SZP;
  _Float16* v16 = w + 2 * SZP;
  _Float16* qT  = w + 3 * SZP;
  _Float16* kT  = w + 4 * SZP;
  _Float16* vT  = w + 5 * SZP;
  _Float16* Sr  = w + 6 * SZP;
  _Float16* Sl  = Sr + SSZ;
  _Float16* xr  = Sl + SSZ;
  _Float16* xl  = xr + SZP;

  const dim3 blk(256);
  // 1) projections (write both [x][y] and [y][x] layouts for the l-path GEMMs)
  proj_qkv_kernel<<<ROWS / 16, blk, 0, stream>>>(query, Wq, q16, qT);
  proj_qkv_kernel<<<ROWS / 16, blk, 0, stream>>>(key,   Wk, k16, kT);
  proj_qkv_kernel<<<ROWS / 16, blk, 0, stream>>>(value, Wv, v16, vT);
  // 2) masked scores: row mode per (x,h), column mode per (y,h)
  scores_kernel<<<dim3(Nn, Hh), blk, 0, stream>>>(q16, k16, mask, Sr, 0);
  scores_kernel<<<dim3(Nn, Hh), blk, 0, stream>>>(qT,  kT,  mask, Sl, 1);
  // 3) joint softmax over concat(r,l), in place -> att
  softmax_kernel<<<(Nn * Nn * Hh) / 8, blk, 0, stream>>>(Sr, Sl);
  // 4) att @ V for both paths
  av_kernel<<<dim3(Nn, Hh), blk, 0, stream>>>(Sr, v16, xr);
  av_kernel<<<dim3(Nn, Hh), blk, 0, stream>>>(Sl, vT,  xl);
  // 5) (x_r + x_l) @ Wo^T -> f32 output
  out_kernel<<<ROWS / 16, blk, 0, stream>>>(xr, xl, Wo, out);
  (void)in_sizes; (void)n_in; (void)out_size; (void)ws_size;
}
